// Local2TransformerEncoder_2740189134929
// MI455X (gfx1250) — matrix operand order repaired and checked by reading, not compiled, not run on hardware
//
#include <hip/hip_runtime.h>
#include <hip/hip_bf16.h>
#include <math.h>

typedef __attribute__((ext_vector_type(16))) _Float16 v16h;
typedef __attribute__((ext_vector_type(8)))  _Float16 v8h;
typedef __attribute__((ext_vector_type(8)))  float    v8f;

#define CB    4
#define CT    4096
#define CD    512
#define CH    8
#define CHD   64
#define CMLP  2048
#define NTOK  (CB * CT)          // 16384
#define LN_EPS 1e-6f

// ---------------------------------------------------------------------------
// WMMA helpers (gfx1250 wave32, f32 <- f16 x f16, 16x16x32)
// ---------------------------------------------------------------------------
__device__ __forceinline__ v8f wmma32(v16h a, v16h b, v8f c) {
  return __builtin_amdgcn_wmma_f32_16x16x32_f16(
      /*neg_a=*/false, a, /*neg_b=*/false, b,
      /*c_mod=*/(short)0, c, /*reuse_a=*/false, /*reuse_b=*/false);
}

// A-matrix fragment (16x32 f16). Per ISA: lane group g = lane>>4, r = lane&15.
// element e<8  -> k = g*8 + e          (contiguous 8 halves at +g*8)
// element e>=8 -> k = 16 + g*8 + (e-8) (contiguous 8 halves at +16+g*8)
__device__ __forceinline__ v16h frag_a_ld(const _Float16* base, int strideHalves) {
  const int lane = threadIdx.x & 31;
  const int g = lane >> 4, r = lane & 15;
  const _Float16* p = base + (size_t)r * strideHalves + g * 8;
  union { v16h v; v8h h[2]; } u;
  u.h[0] = *(const v8h*)p;
  u.h[1] = *(const v8h*)(p + 16);
  return u.v;
}

// B-matrix fragment (32x16 f16) from B^T storage [n][k].
// element e<8 -> k = g*8 + e; e>=8 -> k = 16 + g*8 + (e-8) (same as A).
__device__ __forceinline__ v16h frag_b_ld(const _Float16* base, int strideHalves) {
  const int lane = threadIdx.x & 31;
  const int g = lane >> 4, r = lane & 15;
  const _Float16* p = base + (size_t)r * strideHalves + g * 8;
  union { v16h v; v8h h[2]; } u;
  u.h[0] = *(const v8h*)p;
  u.h[1] = *(const v8h*)(p + 16);
  return u.v;
}

// Guarded per-lane-row versions (rowp already includes the per-lane row).
__device__ __forceinline__ v16h frag_a_row(const _Float16* rowp, bool valid) {
  const int g = (threadIdx.x >> 4) & 1;
  v8h z = {0, 0, 0, 0, 0, 0, 0, 0};
  union { v16h v; v8h h[2]; } u;
  u.h[0] = valid ? *(const v8h*)(rowp + g * 8) : z;
  u.h[1] = valid ? *(const v8h*)(rowp + 16 + g * 8) : z;
  return u.v;
}
__device__ __forceinline__ v16h frag_b_row(const _Float16* rowp, bool valid) {
  const int g = (threadIdx.x >> 4) & 1;
  v8h z = {0, 0, 0, 0, 0, 0, 0, 0};
  union { v16h v; v8h h[2]; } u;
  u.h[0] = valid ? *(const v8h*)(rowp + g * 8) : z;
  u.h[1] = valid ? *(const v8h*)(rowp + 16 + g * 8) : z;
  return u.v;
}

// ---------------------------------------------------------------------------
// Weight convert + transpose: src f32 [l][K][N] -> dst f16 [l][rowOff+n][k]
// ---------------------------------------------------------------------------
__global__ void wconv_kernel(const float* __restrict__ src, _Float16* __restrict__ dst,
                             int K, int N, long dstLStride, int dstRowOff, long total) {
  long i = (long)blockIdx.x * blockDim.x + threadIdx.x;
  if (i >= total) return;
  long kn = (long)K * N;
  int l = (int)(i / kn);
  long rem = i - (long)l * kn;
  int n = (int)(rem / K);
  int k = (int)(rem - (long)n * K);
  dst[l * dstLStride + (long)(dstRowOff + n) * K + k] =
      (_Float16)src[((long)l * K + k) * N + n];
}

// ---------------------------------------------------------------------------
// Embedding + sinusoidal PE (f32)
// ---------------------------------------------------------------------------
__global__ void embed_kernel(const int* __restrict__ toks, const float* __restrict__ emb,
                             float* __restrict__ x) {
  const int t = blockIdx.x;              // 0 .. NTOK-1
  const int pos = t % CT;
  const int tok = toks[t];
  const float kdiv = -2.0f * 9.210340371976184f / (float)CD;   // -2*ln(10000)/D
  for (int c = threadIdx.x; c < CD; c += blockDim.x) {
    float pe;
    if (c < CD / 2) pe = sinf((float)pos * expf((float)c * kdiv));
    else            pe = cosf((float)pos * expf((float)(c - CD / 2) * kdiv));
    x[(size_t)t * CD + c] = emb[(size_t)tok * CD + c] + pe;
  }
}

// ---------------------------------------------------------------------------
// LayerNorm over D=512; one wave per row; 8 rows per 256-thread block
// ---------------------------------------------------------------------------
template <bool OUT_F16>
__global__ __launch_bounds__(256) void ln_kernel(const float* __restrict__ x,
                                                 const float* __restrict__ sc,
                                                 const float* __restrict__ bi,
                                                 void* __restrict__ out) {
  const int wave = threadIdx.x >> 5, lane = threadIdx.x & 31;
  const int row = blockIdx.x * 8 + wave;
  const float* xp = x + (size_t)row * CD;
  float v[16];
  float s = 0.f;
#pragma unroll
  for (int i = 0; i < 16; ++i) { v[i] = xp[lane + i * 32]; s += v[i]; }
#pragma unroll
  for (int m = 1; m < 32; m <<= 1) s += __shfl_xor(s, m, 32);
  const float mean = s * (1.f / CD);
  float var = 0.f;
#pragma unroll
  for (int i = 0; i < 16; ++i) { float d = v[i] - mean; var += d * d; }
#pragma unroll
  for (int m = 1; m < 32; m <<= 1) var += __shfl_xor(var, m, 32);
  const float rs = rsqrtf(var * (1.f / CD) + LN_EPS);
#pragma unroll
  for (int i = 0; i < 16; ++i) {
    int c = lane + i * 32;
    float y = (v[i] - mean) * rs * sc[c] + bi[c];
    if (OUT_F16) ((_Float16*)out)[(size_t)row * CD + c] = (_Float16)y;
    else         ((float*)out)[(size_t)row * CD + c] = y;
  }
}

// ---------------------------------------------------------------------------
// WMMA GEMM: out[M,N] = act(A[M,K](f16) * Wt[N,K]^T(f16) + bias) (+resid f32)
// Block 256 = 8 waves (4x2); block tile 128x128; wave tile 32x64; K step 32.
// Double-buffered LDS (ping-pong): next tile's global loads are issued before
// the current tile's WMMAs, stored to the alternate buffer afterwards, with a
// single barrier per iteration.
// ---------------------------------------------------------------------------
#define LSTR 40   // LDS row stride in halves for a 32-wide k tile

template <bool HAS_BIAS, bool RELU, bool RESID, bool OUT_F16>
__global__ __launch_bounds__(256) void gemm_kernel(
    const _Float16* __restrict__ A, const _Float16* __restrict__ Wt,
    const float* __restrict__ bias, const float* __restrict__ resid,
    void* __restrict__ out, int N, int K) {
  __shared__ _Float16 As[2][128 * LSTR];
  __shared__ _Float16 Bs[2][128 * LSTR];
  const int tid = threadIdx.x;
  const int lane = tid & 31, wave = tid >> 5;
  const int wm = wave >> 1, wn = wave & 1;
  const int m0 = blockIdx.y * 128, n0 = blockIdx.x * 128;
  const int lrow = tid >> 1, lseg = (tid & 1) * 16;
  const _Float16* gArow = A + (size_t)(m0 + lrow) * K + lseg;
  const _Float16* gBrow = Wt + (size_t)(n0 + lrow) * K + lseg;

  v8f acc[2][4];
#pragma unroll
  for (int i = 0; i < 2; ++i)
#pragma unroll
    for (int j = 0; j < 4; ++j) acc[i][j] = (v8f){0, 0, 0, 0, 0, 0, 0, 0};

  // preload tile 0 into LDS buffer 0
  {
    const uint4* ga = (const uint4*)gArow;
    const uint4* gb = (const uint4*)gBrow;
    uint4 a0 = ga[0], a1 = ga[1], b0 = gb[0], b1 = gb[1];
    *(uint4*)&As[0][lrow * LSTR + lseg]     = a0;
    *(uint4*)&As[0][lrow * LSTR + lseg + 8] = a1;
    *(uint4*)&Bs[0][lrow * LSTR + lseg]     = b0;
    *(uint4*)&Bs[0][lrow * LSTR + lseg + 8] = b1;
  }
  __syncthreads();

  const int nk = K >> 5;
  for (int it = 0; it < nk; ++it) {
    const int cur = it & 1, nxt = cur ^ 1;
    const bool has_next = (it + 1) < nk;

    // issue next tile's global loads early (latency hidden by WMMAs below)
    uint4 na0, na1, nb0, nb1;
    if (has_next) {
      const uint4* ga = (const uint4*)(gArow + (it + 1) * 32);
      const uint4* gb = (const uint4*)(gBrow + (it + 1) * 32);
      na0 = ga[0]; na1 = ga[1]; nb0 = gb[0]; nb1 = gb[1];
      if (it + 2 < nk) {
        __builtin_prefetch(gArow + (it + 2) * 32, 0, 1);
        __builtin_prefetch(gBrow + (it + 2) * 32, 0, 1);
      }
    }

    // compute from current LDS buffer
    v16h af[2], bf[4];
#pragma unroll
    for (int mi = 0; mi < 2; ++mi)
      af[mi] = frag_a_ld(&As[cur][(wm * 32 + mi * 16) * LSTR], LSTR);
#pragma unroll
    for (int ni = 0; ni < 4; ++ni)
      bf[ni] = frag_b_ld(&Bs[cur][(wn * 64 + ni * 16) * LSTR], LSTR);
#pragma unroll
    for (int mi = 0; mi < 2; ++mi)
#pragma unroll
      for (int ni = 0; ni < 4; ++ni)
        acc[mi][ni] = wmma32(af[mi], bf[ni], acc[mi][ni]);

    // stage next tile into the alternate buffer, then one barrier
    if (has_next) {
      *(uint4*)&As[nxt][lrow * LSTR + lseg]     = na0;
      *(uint4*)&As[nxt][lrow * LSTR + lseg + 8] = na1;
      *(uint4*)&Bs[nxt][lrow * LSTR + lseg]     = nb0;
      *(uint4*)&Bs[nxt][lrow * LSTR + lseg + 8] = nb1;
      __syncthreads();
    }
  }

  const int g = lane >> 4, c = lane & 15;
#pragma unroll
  for (int mi = 0; mi < 2; ++mi)
#pragma unroll
    for (int ni = 0; ni < 4; ++ni) {
      const int col = n0 + wn * 64 + ni * 16 + c;
      float bv = HAS_BIAS ? bias[col] : 0.f;
#pragma unroll
      for (int e = 0; e < 8; ++e) {
        const int row = m0 + wm * 32 + mi * 16 + e + 8 * g;
        float v = acc[mi][ni][e] + bv;
        if (RELU) v = fmaxf(v, 0.f);
        if (RESID) v += resid[(size_t)row * N + col];
        if (OUT_F16) ((_Float16*)out)[(size_t)row * N + col] = (_Float16)v;
        else         ((float*)out)[(size_t)row * N + col] = v;
      }
    }
}

// ---------------------------------------------------------------------------
// Local block attention. Block = 64 threads = 2 waves = 2 heads.
// Grid: (nb, H/2, B). One 64x64 token block per x-index (staggered by 32 on
// odd layers). qkv layout: [B*T][1536] f16, cols 0:Q 512:K 1024:V.
// ---------------------------------------------------------------------------
#define PPAD 72

__global__ __launch_bounds__(64) void attn_kernel(
    const _Float16* __restrict__ qkv, const int* __restrict__ toks,
    _Float16* __restrict__ aout, int stagger) {
  __shared__ _Float16 Vt[2][64][PPAD];   // V transposed: [hd][key]
  __shared__ _Float16 Ps[2][64][PPAD];   // probs:        [q][key]
  __shared__ float    nbias[2][64];      // key mask bias
  const int w = threadIdx.x >> 5, lane = threadIdx.x & 31;
  const int head = blockIdx.y * 2 + w;
  const int b = blockIdx.z;
  const int t0 = blockIdx.x * 64 - (stagger ? 32 : 0);
  const long bT = (long)b * CT;
  const int colQ = head * CHD, colK = CD + head * CHD, colV = 2 * CD + head * CHD;

  // ---- stage V transposed into LDS + key mask bias ----
  for (int key2 = 0; key2 < 64; key2 += 16) {
    const int key = key2 + (lane & 15);
    const int dbase = (lane >> 4) * 32;
    const int t = t0 + key;
    const bool valid = (t >= 0) && (t < CT);
    const int tc = valid ? t : 0;
    const _Float16* vp = qkv + (bT + tc) * 1536 + colV + dbase;
#pragma unroll
    for (int dd = 0; dd < 32; ++dd) {
      _Float16 xv = valid ? vp[dd] : (_Float16)0.0f;
      Vt[w][dbase + dd][key] = xv;
    }
  }
  for (int key = lane; key < 64; key += 32) {
    const int t = t0 + key;
    const bool inr = (t >= 0) && (t < CT);
    const int tc = inr ? t : 0;
    nbias[w][key] = (inr && (toks[bT + tc] > 0)) ? 0.f : -1.0e9f;
  }

  // ---- K as B fragments (B^T rows = K rows, natural [key][hd] layout) ----
  v16h bK[4][2];
#pragma unroll
  for (int nt = 0; nt < 4; ++nt)
#pragma unroll
    for (int ks = 0; ks < 2; ++ks) {
      const int r = lane & 15;
      const int tk = t0 + nt * 16 + r;
      const bool valid = (tk >= 0) && (tk < CT);
      const int tc = valid ? tk : 0;
      bK[nt][ks] = frag_b_row(qkv + (bT + tc) * 1536 + colK + ks * 32, valid);
    }
  __syncthreads();

  // ---- S = (Q*scale) K^T, mask, softmax -> Ps (f16, A-layout-friendly) ----
  for (int mt = 0; mt < 4; ++mt) {
    v8f s[4];
#pragma unroll
    for (int nt = 0; nt < 4; ++nt) s[nt] = (v8f){0, 0, 0, 0, 0, 0, 0, 0};
#pragma unroll
    for (int ks = 0; ks < 2; ++ks) {
      const int r = lane & 15;
      const int tq = t0 + mt * 16 + r;
      const bool valid = (tq >= 0) && (tq < CT);
      const int tc = valid ? tq : 0;
      v16h aQ = frag_a_row(qkv + (bT + tc) * 1536 + colQ + ks * 32, valid);
#pragma unroll
      for (int nt = 0; nt < 4; ++nt) s[nt] = wmma32(aQ, bK[nt][ks], s[nt]);
    }
    const int g = lane >> 4, c = lane & 15;
#pragma unroll
    for (int e = 0; e < 8; ++e) {   // one matrix row per e (m = mt*16 + e + 8g)
      float vals[4];
      float mx = -3.0e38f;
#pragma unroll
      for (int nt = 0; nt < 4; ++nt) {
        vals[nt] = s[nt][e] * 0.125f + nbias[w][nt * 16 + c];
        mx = fmaxf(mx, vals[nt]);
      }
#pragma unroll
      for (int m = 1; m < 16; m <<= 1) mx = fmaxf(mx, __shfl_xor(mx, m, 32));
      float sum = 0.f;
#pragma unroll
      for (int nt = 0; nt < 4; ++nt) { vals[nt] = __expf(vals[nt] - mx); sum += vals[nt]; }
#pragma unroll
      for (int m = 1; m < 16; m <<= 1) sum += __shfl_xor(sum, m, 32);
      const float inv = 1.f / sum;
      const int mrow = mt * 16 + e + 8 * g;
#pragma unroll
      for (int nt = 0; nt < 4; ++nt)
        Ps[w][mrow][nt * 16 + c] = (_Float16)(vals[nt] * inv);
    }
  }
  __syncthreads();

  // ---- O = P x V  (B fragments from transposed V in LDS) ----
  v16h bV[4][2];
#pragma unroll
  for (int nt = 0; nt < 4; ++nt)
#pragma unroll
    for (int ks = 0; ks < 2; ++ks)
      bV[nt][ks] = frag_b_ld(&Vt[w][nt * 16][ks * 32], PPAD);

  for (int mt = 0; mt < 4; ++mt) {
    v8f o[4];
#pragma unroll
    for (int nt = 0; nt < 4; ++nt) o[nt] = (v8f){0, 0, 0, 0, 0, 0, 0, 0};
#pragma unroll
    for (int ks = 0; ks < 2; ++ks) {
      v16h aP = frag_a_ld(&Ps[w][mt * 16][ks * 32], PPAD);
#pragma unroll
      for (int nt = 0; nt < 4; ++nt) o[nt] = wmma32(aP, bV[nt][ks], o[nt]);
    }
    const int g = lane >> 4, c = lane & 15;
#pragma unroll
    for (int e = 0; e < 8; ++e) {
      const int t = t0 + mt * 16 + e + 8 * g;
      if (t >= 0 && t < CT) {
#pragma unroll
        for (int nt = 0; nt < 4; ++nt)
          aout[(bT + t) * CD + head * CHD + nt * 16 + c] = (_Float16)o[nt][e];
      }
    }
  }
}

// ---------------------------------------------------------------------------
// Host-side launch
// ---------------------------------------------------------------------------
extern "C" void kernel_launch(void* const* d_in, const int* in_sizes, int n_in,
                              void* d_out, int out_size, void* d_ws, size_t ws_size,
                              hipStream_t stream) {
  (void)in_sizes; (void)n_in; (void)out_size; (void)ws_size;
  const int*   toks  = (const int*)d_in[0];
  const float* emb   = (const float*)d_in[1];
  const float* wq    = (const float*)d_in[2];
  const float* wk    = (const float*)d_in[3];
  const float* wv    = (const float*)d_in[4];
  const float* wo    = (const float*)d_in[5];
  const float* ln1_s = (const float*)d_in[6];
  const float* ln1_b = (const float*)d_in[7];
  const float* ln2_s = (const float*)d_in[8];
  const float* ln2_b = (const float*)d_in[9];
  const float* w1    = (const float*)d_in[10];
  const float* b1    = (const float*)d_in[11];
  const float* w2    = (const float*)d_in[12];
  const float* b2    = (const float*)d_in[13];
  const float* lnf_s = (const float*)d_in[14];
  const float* lnf_b = (const float*)d_in[15];

  char* ws = (char*)d_ws;
  size_t off = 0;
  auto alloc = [&](size_t bytes) { size_t o = off; off += (bytes + 255) & ~(size_t)255; return o; };
  _Float16* wqkvT = (_Float16*)(ws + alloc((size_t)6 * 1536 * 512 * 2));
  _Float16* woT   = (_Float16*)(ws + alloc((size_t)6 * 512 * 512 * 2));
  _Float16* w1T   = (_Float16*)(ws + alloc((size_t)6 * 2048 * 512 * 2));
  _Float16* w2T   = (_Float16*)(ws + alloc((size_t)6 * 512 * 2048 * 2));
  float*    x     = (float*)   (ws + alloc((size_t)NTOK * CD * 4));
  _Float16* h     = (_Float16*)(ws + alloc((size_t)NTOK * CD * 2));
  _Float16* qkv   = (_Float16*)(ws + alloc((size_t)NTOK * 1536 * 2));
  _Float16* attno = (_Float16*)(ws + alloc((size_t)NTOK * CD * 2));
  _Float16* mid   = (_Float16*)(ws + alloc((size_t)NTOK * CMLP * 2));

  // -- weight prep (f32 -> f16, transposed to [N][K]) --
  {
    long tot = 6L * 512 * 512;
    int blocks = (int)((tot + 255) / 256);
    wconv_kernel<<<blocks, 256, 0, stream>>>(wq, wqkvT, 512, 512, 1536L * 512, 0, tot);
    wconv_kernel<<<blocks, 256, 0, stream>>>(wk, wqkvT, 512, 512, 1536L * 512, 512, tot);
    wconv_kernel<<<blocks, 256, 0, stream>>>(wv, wqkvT, 512, 512, 1536L * 512, 1024, tot);
    wconv_kernel<<<blocks, 256, 0, stream>>>(wo, woT, 512, 512, 512L * 512, 0, tot);
    long tot2 = 6L * 512 * 2048;
    int blocks2 = (int)((tot2 + 255) / 256);
    wconv_kernel<<<blocks2, 256, 0, stream>>>(w1, w1T, 512, 2048, 2048L * 512, 0, tot2);
    wconv_kernel<<<blocks2, 256, 0, stream>>>(w2, w2T, 2048, 512, 512L * 2048, 0, tot2);
  }

  // -- embedding + positional encoding --
  embed_kernel<<<NTOK, 256, 0, stream>>>(toks, emb, x);

  const dim3 gLN(NTOK / 8), bLN(256);
  for (int l = 0; l < 6; ++l) {
    const int stag = l & 1;
    const int nb = stag ? (CT / 64 + 1) : (CT / 64);
    // LN1 -> h (f16)
    ln_kernel<true><<<gLN, bLN, 0, stream>>>(x, ln1_s + l * CD, ln1_b + l * CD, h);
    // QKV projection: [NTOK,512] x [512,1536]
    gemm_kernel<false, false, false, true>
        <<<dim3(1536 / 128, NTOK / 128), 256, 0, stream>>>(
            h, wqkvT + (size_t)l * 1536 * 512, nullptr, nullptr, qkv, 1536, 512);
    // local block attention
    attn_kernel<<<dim3(nb, CH / 2, CB), 64, 0, stream>>>(qkv, toks, attno, stag);
    // output projection + residual -> x (f32)
    gemm_kernel<false, false, true, false>
        <<<dim3(512 / 128, NTOK / 128), 256, 0, stream>>>(
            attno, woT + (size_t)l * 512 * 512, nullptr, x, x, 512, 512);
    // LN2 -> h (f16)
    ln_kernel<true><<<gLN, bLN, 0, stream>>>(x, ln2_s + l * CD, ln2_b + l * CD, h);
    // MLP1: relu(h*W1 + b1) -> mid (f16)
    gemm_kernel<true, true, false, true>
        <<<dim3(CMLP / 128, NTOK / 128), 256, 0, stream>>>(
            h, w1T + (size_t)l * CMLP * 512, b1 + l * CMLP, nullptr, mid, CMLP, 512);
    // MLP2: mid*W2 + b2 + residual -> x (f32)
    gemm_kernel<true, false, true, false>
        <<<dim3(512 / 128, NTOK / 128), 256, 0, stream>>>(
            mid, w2T + (size_t)l * 512 * CMLP, b2 + l * CD, x, x, 512, CMLP);
  }
  // final LN -> d_out (f32)
  ln_kernel<false><<<gLN, bLN, 0, stream>>>(x, lnf_s, lnf_b, d_out);
}
